// MHSA_11209864642921
// MI455X (gfx1250) — matrix operand order repaired and checked by reading, not compiled, not run on hardware
//
#include <hip/hip_runtime.h>
#include <hip/hip_bf16.h>
#include <stdint.h>

// ---------------------------------------------------------------------------
// MHSA on gfx1250: fp16 WMMA (v_wmma_f32_16x16x32_f16) for all matmuls,
// TDM (tensor_load_to_lds) double-buffered GEMM staging, fused flash
// attention. B=2, N=2048, E=1024, H=16, D=64.
// ---------------------------------------------------------------------------

typedef _Float16 v16h __attribute__((ext_vector_type(16)));
typedef float    v8f  __attribute__((ext_vector_type(8)));
typedef uint32_t v4u  __attribute__((ext_vector_type(4)));
typedef int      v8i  __attribute__((ext_vector_type(8)));
typedef int      v4i  __attribute__((ext_vector_type(4)));

union Frag { uint32_t u[8]; v16h h; };

#define SEQ   2048
#define EMB   1024
#define HEADS 16
#define HDIM  64
#define MTOK  4096   // B * SEQ
#define LDSTR 40     // padded LDS row stride in fp16 elems (32 data + 4-DWORD pad)

static __device__ __forceinline__ v8f wmma_f16(const Frag& a, const Frag& b, v8f c) {
    return __builtin_amdgcn_wmma_f32_16x16x32_f16(
        /*neg_a=*/false, a.h, /*neg_b=*/false, b.h,
        /*c_mod=*/(short)0, c, /*reuse_a=*/false, /*reuse_b=*/false);
}

// A-fragment pair offset (16-bit A 16x32): lane<16 -> K {0..15},
// lane>=16 -> K {16..31}; VGPR j holds the pair starting at:
static __device__ __forceinline__ int a_pair(int lane, int j) {
    return ((lane < 16) ? 0 : 16) + 2 * j;
}
// B-fragment pair offset (16-bit B 32x16): VGPR j: lanes0-15 K=2j, lanes16-31 K=16+2j
static __device__ __forceinline__ int b_pair(int lane, int j) {
    return ((lane < 16) ? 0 : 16) + 2 * j;
}

// ---------------------------------------------------------------------------
// Tensor Data Mover: DMA a (tile_h x 32) fp16 tile out of a (tensor_h x 1024)
// row-major fp16 tensor into LDS, inserting a 4-DWORD pad every 16 DWORDs
// (i.e. LDS row stride = 40 elems, matching the fragment-read layout).
// D# bitfields per CDNA5 ISA ch.8. Groups 2/3 zero (2D tensor).
// This toolchain's builtin takes 6 args (g0, g1, g2, g3, g4, cpol).
static __device__ __forceinline__ void tdm_load_tile(uint32_t lds_off, const void* gptr,
                                                     uint32_t tensor_h, uint32_t tile_h) {
    const uint64_t ga = (uint64_t)gptr;
    const v4u g0 = { 1u,                                   // count=1, user mode
                     lds_off,                              // lds_addr
                     (uint32_t)ga,                         // global_addr[31:0]
                     ((uint32_t)(ga >> 32) & 0x01FFFFFFu)  // global_addr[56:32]
                     | (2u << 30) };                       // type=2 ("image")
    const v8i g1 = {
        (int)((1u << 16)      // data_size = 2 bytes
            | (1u << 20)      // pad_enable
            | (3u << 22)      // pad_interval: every 16 DWORDs (64B = one 32-elem row)
            | (3u << 25)),    // pad_amount: 4 DWORDs (stride 32 -> 40 elems)
        (int)((EMB & 0xFFFFu) << 16),                      // tensor_dim0 lo16 (atomic_barrier=0)
        (int)((EMB >> 16) | ((tensor_h & 0xFFFFu) << 16)), // tensor_dim0 hi16 | tensor_dim1 lo16
        (int)((tensor_h >> 16) | (32u << 16)),             // tensor_dim1 hi16 | tile_dim0 = 32
        (int)tile_h,                                       // tile_dim1 (tile_dim2 = 0)
        (int)EMB,                                          // tensor_dim0_stride lo32
        0, 0 };                                            // dim0_stride hi | dim1_stride = 0
    const v4i z4 = { 0, 0, 0, 0 };
    const v8i z8 = { 0, 0, 0, 0, 0, 0, 0, 0 };
    __builtin_amdgcn_tensor_load_to_lds(g0, g1, z4, z4, z8, 0);
}

static __device__ __forceinline__ uint32_t lds_offset(const void* p) {
    return (uint32_t)(uintptr_t)p;   // generic LDS pointer: low 32 bits = LDS offset
}

// ---------------------------------------------------------------------------
__global__ void cvt_f32_f16(const float* __restrict__ s, _Float16* __restrict__ d, int n) {
    int i = blockIdx.x * blockDim.x + threadIdx.x;
    if (i < n) d[i] = (_Float16)s[i];
}

// ---------------------------------------------------------------------------
// One K-step of the block GEMM: 4 WMMAs per wave from LDS-staged tiles.
static __device__ __forceinline__ void gemm_step(const _Float16* __restrict__ As,
                                                 const _Float16* __restrict__ Bs,
                                                 int lane, int wr, int wc, v8f* acc) {
    Frag a;
    const _Float16* Ap = As + (wr * 16 + (lane & 15)) * LDSTR;
#pragma unroll
    for (int j = 0; j < 8; ++j)
        a.u[j] = *(const uint32_t*)(Ap + a_pair(lane, j));
#pragma unroll
    for (int t = 0; t < 4; ++t) {
        Frag b;
        const _Float16* Bp = Bs + (wc * 64 + t * 16 + (lane & 15)) * LDSTR;
#pragma unroll
        for (int j = 0; j < 8; ++j)
            b.u[j] = *(const uint32_t*)(Bp + b_pair(lane, j));
        acc[t] = wmma_f16(a, b, acc[t]);
    }
}

// GEMM: out = A[M,1024] @ W[1024,1024]^T + bias, fp16 in, f32 accumulate.
// Block tile 64x128, 256 threads = 8 waves (4x2), wave tile 16x64.
// Operand staging: TDM double-buffered (wave 0 issues, TENSORcnt-synced).
// MODE 0: f32 [row*1024+col]; 1: Q fp16 [B,H,N,D] * 8; 2: K fp16 [B,H,N,D];
// 3: V fp16 [B,H,D,N].
template <int MODE>
__global__ __launch_bounds__(256) void gemm16(const _Float16* __restrict__ A,
                                              const _Float16* __restrict__ W,
                                              const float* __restrict__ bias,
                                              void* __restrict__ out) {
    __shared__ _Float16 AsBuf[2][64 * LDSTR];
    __shared__ _Float16 BsBuf[2][128 * LDSTR];

    const int tid  = threadIdx.x;
    const int lane = tid & 31;
    const int wv   = tid >> 5;
    const int wr   = wv >> 1;          // 0..3 (M)
    const int wc   = wv & 1;           // 0..1 (N)
    const int rowBase = blockIdx.y * 64;
    const int colBase = blockIdx.x * 128;

    const v8f vzero = {0.f, 0.f, 0.f, 0.f, 0.f, 0.f, 0.f, 0.f};
    v8f acc[4] = {vzero, vzero, vzero, vzero};

    // Prologue: DMA first tile pair into buffer 0.
    if (wv == 0) {
        tdm_load_tile(lds_offset(&AsBuf[0][0]), A + (size_t)rowBase * EMB, MTOK, 64);
        tdm_load_tile(lds_offset(&BsBuf[0][0]), W + (size_t)colBase * EMB, EMB, 128);
    }

#pragma unroll 1
    for (int ks = 0; ks < EMB - 32; ks += 32) {
        const int cur = (ks >> 5) & 1;
        if (wv == 0) {
            // Prefetch next tile pair into the other buffer, then wait for the
            // current pair (<=2 outstanding keeps only the new pair in flight).
            tdm_load_tile(lds_offset(&AsBuf[cur ^ 1][0]),
                          A + (size_t)rowBase * EMB + (ks + 32), MTOK, 64);
            tdm_load_tile(lds_offset(&BsBuf[cur ^ 1][0]),
                          W + (size_t)colBase * EMB + (ks + 32), EMB, 128);
            __builtin_amdgcn_s_wait_tensorcnt(2);
        }
        __syncthreads();
        gemm_step(AsBuf[cur], BsBuf[cur], lane, wr, wc, acc);
        __syncthreads();
    }
    {   // Peeled last K-step (immediate wait value must be a constant).
        if (wv == 0) __builtin_amdgcn_s_wait_tensorcnt(0);
        __syncthreads();
        gemm_step(AsBuf[1], BsBuf[1], lane, wr, wc, acc);
    }

    // Epilogue: bias + layout-specific store.
#pragma unroll
    for (int t = 0; t < 4; ++t) {
        const int col = colBase + wc * 64 + t * 16 + (lane & 15);
        const float bvv = bias[col];
#pragma unroll
        for (int r = 0; r < 8; ++r) {
            const int row = rowBase + wr * 16 + r + ((lane < 16) ? 0 : 8);
            float v = acc[t][r] + bvv;
            if (MODE == 0) {
                ((float*)out)[(size_t)row * EMB + col] = v;
            } else {
                const int b = row >> 11, nn = row & 2047;
                const int h = col >> 6,  d  = col & 63;
                _Float16* o16 = (_Float16*)out;
                if (MODE == 1)
                    o16[((size_t)(b * HEADS + h) * SEQ + nn) * HDIM + d] = (_Float16)(v * 8.0f);
                else if (MODE == 2)
                    o16[((size_t)(b * HEADS + h) * SEQ + nn) * HDIM + d] = (_Float16)v;
                else
                    o16[((size_t)(b * HEADS + h) * HDIM + d) * SEQ + nn] = (_Float16)v;
            }
        }
    }
}

// ---------------------------------------------------------------------------
// Fused attention (flash-style): grid = B*H*(SEQ/64) blocks, 128 threads.
// Each wave owns 16 query rows; online softmax over 32-key chunks.
__global__ __launch_bounds__(128) void attn(const _Float16* __restrict__ Q,
                                            const _Float16* __restrict__ Km,
                                            const _Float16* __restrict__ Vt,
                                            _Float16* __restrict__ CH) {
    __shared__ _Float16 Plds[4 * 16 * 34];   // per-wave 16x32 P tile, padded stride 34

    const int tid  = threadIdx.x;
    const int lane = tid & 31;
    const int wv   = tid >> 5;
    const int tile = blockIdx.x & 31;        // 32 row tiles of 64
    const int bh   = blockIdx.x >> 5;        // 0..31
    const int b    = bh >> 4, h = bh & 15;
    const int rowBase = tile * 64 + wv * 16;

    const _Float16* Qp = Q  + (size_t)bh * SEQ * HDIM;
    const _Float16* Kp = Km + (size_t)bh * SEQ * HDIM;
    const _Float16* Vp = Vt + (size_t)bh * HDIM * SEQ;
    _Float16* Pw = Plds + wv * 16 * 34;

    // Q A-fragments, K-dim = head dim 64 -> two 16x32 fragments.
    Frag qa[2];
    {
        const _Float16* qrow = Qp + (size_t)(rowBase + (lane & 15)) * HDIM;
#pragma unroll
        for (int half = 0; half < 2; ++half)
#pragma unroll
            for (int j = 0; j < 8; ++j)
                qa[half].u[j] = *(const uint32_t*)(qrow + half * 32 + a_pair(lane, j));
    }

    const v8f vzero = {0.f, 0.f, 0.f, 0.f, 0.f, 0.f, 0.f, 0.f};
    v8f o[4] = {vzero, vzero, vzero, vzero};
    float m[8], l[8];
#pragma unroll
    for (int r = 0; r < 8; ++r) { m[r] = -1e30f; l[r] = 0.f; }

#pragma unroll 1
    for (int kc = 0; kc < SEQ; kc += 32) {
        // Speculative prefetch of the next K/V chunk (global_prefetch_b8);
        // out-of-range prefetches are silently dropped.
        __builtin_prefetch(Kp + (size_t)(kc + 32 + lane) * HDIM, 0, 0);
        __builtin_prefetch(Vp + (size_t)lane * SEQ + kc + 32, 0, 0);
        __builtin_prefetch(Vp + (size_t)(lane + 32) * SEQ + kc + 32, 0, 0);

        // S = Q @ K^T for this 16x32 key chunk (scale already folded into Q).
        v8f s0 = vzero, s1 = vzero;
#pragma unroll
        for (int t = 0; t < 2; ++t) {
#pragma unroll
            for (int half = 0; half < 2; ++half) {
                Frag kb;
                const int key = kc + t * 16 + (lane & 15);
#pragma unroll
                for (int j = 0; j < 8; ++j) {
                    const int d = half * 32 + b_pair(lane, j);
                    kb.u[j] = *(const uint32_t*)(Kp + (size_t)key * HDIM + d);
                }
                if (t == 0) s0 = wmma_f16(qa[half], kb, s0);
                else        s1 = wmma_f16(qa[half], kb, s1);
            }
        }

        // Online softmax. C layout: VGPR r -> row r (lanes 0-15) / row r+8
        // (lanes 16-31); xor-shuffles with mask<=8 reduce within each half.
#pragma unroll
        for (int r = 0; r < 8; ++r) {
            float mx = fmaxf(s0[r], s1[r]);
#pragma unroll
            for (int msk = 8; msk >= 1; msk >>= 1)
                mx = fmaxf(mx, __shfl_xor(mx, msk, 32));
            const float mn = fmaxf(m[r], mx);
            const float sc = __expf(m[r] - mn);
            const float p0 = __expf(s0[r] - mn);
            const float p1 = __expf(s1[r] - mn);
            float rs = p0 + p1;
#pragma unroll
            for (int msk = 8; msk >= 1; msk >>= 1)
                rs += __shfl_xor(rs, msk, 32);
            l[r] = l[r] * sc + rs;
            m[r] = mn;
            s0[r] = p0; s1[r] = p1;
            o[0][r] *= sc; o[1][r] *= sc; o[2][r] *= sc; o[3][r] *= sc;
        }

        // Re-shape P (C layout) -> A-fragment layout via per-wave LDS tile.
        {
            const int prow = (lane < 16) ? 0 : 8;
            const int pc   = lane & 15;
#pragma unroll
            for (int r = 0; r < 8; ++r) {
                Pw[(prow + r) * 34 + pc]      = (_Float16)s0[r];
                Pw[(prow + r) * 34 + 16 + pc] = (_Float16)s1[r];
            }
        }
        asm volatile("s_wait_dscnt 0" ::: "memory");   // CDNA5 split DS counter wait

        Frag pa;
        {
            const _Float16* pr = Pw + (lane & 15) * 34;
#pragma unroll
            for (int j = 0; j < 8; ++j)
                pa.u[j] = *(const uint32_t*)(pr + a_pair(lane, j));
        }

        // O += P @ V (V stored [d][n] so key-pairs are contiguous).
#pragma unroll
        for (int t = 0; t < 4; ++t) {
            Frag vb;
            const int dcol = t * 16 + (lane & 15);
#pragma unroll
            for (int j = 0; j < 8; ++j) {
                const int kk = b_pair(lane, j);
                vb.u[j] = *(const uint32_t*)(Vp + (size_t)dcol * SEQ + kc + kk);
            }
            o[t] = wmma_f16(pa, vb, o[t]);
        }
    }

    // Normalize and store to concat layout [B, N, H*D] (fp16 for final GEMM).
#pragma unroll
    for (int r = 0; r < 8; ++r) l[r] = 1.0f / l[r];
#pragma unroll
    for (int t = 0; t < 4; ++t) {
        const int dcol = t * 16 + (lane & 15);
#pragma unroll
        for (int r = 0; r < 8; ++r) {
            const int qrow = rowBase + r + ((lane < 16) ? 0 : 8);
            const float v = o[t][r] * l[r];
            CH[((size_t)b * SEQ + qrow) * EMB + h * HDIM + dcol] = (_Float16)v;
        }
    }
}

// ---------------------------------------------------------------------------
extern "C" void kernel_launch(void* const* d_in, const int* in_sizes, int n_in,
                              void* d_out, int out_size, void* d_ws, size_t ws_size,
                              hipStream_t stream) {
    (void)in_sizes; (void)n_in; (void)out_size; (void)ws_size;

    const float* x  = (const float*)d_in[0];
    const float* Wq = (const float*)d_in[1];
    const float* bq = (const float*)d_in[2];
    const float* Wk = (const float*)d_in[3];
    const float* bk = (const float*)d_in[4];
    const float* Wv = (const float*)d_in[5];
    const float* bv = (const float*)d_in[6];
    const float* Wo = (const float*)d_in[7];
    const float* bo = (const float*)d_in[8];

    char* ws = (char*)d_ws;
    const size_t MiB = 1ull << 20;
    _Float16* xh  = (_Float16*)(ws + 0 * MiB);    // 8 MiB  (4096x1024)
    _Float16* Wqh = (_Float16*)(ws + 8 * MiB);    // 2 MiB each
    _Float16* Wkh = (_Float16*)(ws + 10 * MiB);
    _Float16* Wvh = (_Float16*)(ws + 12 * MiB);
    _Float16* Woh = (_Float16*)(ws + 14 * MiB);
    _Float16* Qh  = (_Float16*)(ws + 16 * MiB);   // 8 MiB [B,H,N,D], pre-scaled by 8
    _Float16* Kh  = (_Float16*)(ws + 24 * MiB);   // 8 MiB [B,H,N,D]
    _Float16* Vth = (_Float16*)(ws + 32 * MiB);   // 8 MiB [B,H,D,N]
    _Float16* chh = (_Float16*)(ws + 40 * MiB);   // 8 MiB [B,N,E]

    const int nX = MTOK * EMB;     // 4194304
    const int nW = EMB * EMB;      // 1048576
    cvt_f32_f16<<<(nX + 255) / 256, 256, 0, stream>>>(x,  xh,  nX);
    cvt_f32_f16<<<(nW + 255) / 256, 256, 0, stream>>>(Wq, Wqh, nW);
    cvt_f32_f16<<<(nW + 255) / 256, 256, 0, stream>>>(Wk, Wkh, nW);
    cvt_f32_f16<<<(nW + 255) / 256, 256, 0, stream>>>(Wv, Wvh, nW);
    cvt_f32_f16<<<(nW + 255) / 256, 256, 0, stream>>>(Wo, Woh, nW);

    dim3 ggrid(EMB / 128, MTOK / 64);   // 8 x 64
    gemm16<1><<<ggrid, 256, 0, stream>>>(xh, Wqh, bq, (void*)Qh);
    gemm16<2><<<ggrid, 256, 0, stream>>>(xh, Wkh, bk, (void*)Kh);
    gemm16<3><<<ggrid, 256, 0, stream>>>(xh, Wvh, bv, (void*)Vth);

    attn<<<2 * HEADS * (SEQ / 64), 128, 0, stream>>>(Qh, Kh, Vth, chh);

    gemm16<0><<<ggrid, 256, 0, stream>>>(chh, Woh, bo, d_out);
}
